// LorentzRiemannianLayer1Model_24000277250250
// MI455X (gfx1250) — matrix operand order repaired and checked by reading, not compiled, not run on hardware
//
#include <hip/hip_runtime.h>
#include <math.h>

// ---------------- CDNA5 WMMA types ----------------
typedef __attribute__((ext_vector_type(16))) _Float16 v16h;
typedef __attribute__((ext_vector_type(8)))  _Float16 v8h;
typedef __attribute__((ext_vector_type(8)))  float    v8f;

#define EPSF 1e-6f
constexpr int Bc  = 8;
constexpr int Tc  = 2048;
constexpr int Dc  = 64;
constexpr int NHc = 4;
constexpr int DHc = 16;
constexpr int BTc = Bc * Tc;   // 16384 tokens
constexpr int FFc = 4 * Dc;    // 256

__device__ inline v8f wmma16(v16h a, v16h b, v8f c) {
  // D[m][n] = sum_k A[m][k]*B[k][n] + C  (16x16x32, f16 in, f32 acc)
  return __builtin_amdgcn_wmma_f32_16x16x32_f16(false, a, false, b, (short)0, c,
                                                false, false);
}
__device__ inline v16h cat88(v8h a, v8h b) {
  return __builtin_shufflevector(a, b, 0, 1, 2, 3, 4, 5, 6, 7,
                                 8, 9, 10, 11, 12, 13, 14, 15);
}

// ---------------- wave32 cross-lane primitives (VALU path: DPP + permlane) --------
// lane i <-> lane i+16 (identity nibble selects)
__device__ inline unsigned xhalf_u32(unsigned v) {
  return (unsigned)__builtin_amdgcn_permlanex16((int)v, (int)v, 0x76543210,
                                                0xFEDCBA98, false, false);
}
__device__ inline float xhalf_f(float v) {
  return __uint_as_float(xhalf_u32(__float_as_uint(v)));
}
template <int CTRL>
__device__ inline float dpp_add(float v) {
  unsigned u = __float_as_uint(v);
  unsigned o = __builtin_amdgcn_update_dpp(u, u, CTRL, 0xf, 0xf, true);
  return v + __uint_as_float(o);
}
// full-wave (32 lane) all-reduce sum: 4 DPP stages + 1 cross-half permlane
__device__ inline float wsum32(float v) {
  v = dpp_add<0xB1>(v);    // quad_perm(1,0,3,2)  : xor 1
  v = dpp_add<0x4E>(v);    // quad_perm(2,3,0,1)  : xor 2
  v = dpp_add<0x141>(v);   // row_half_mirror     : cross quads in 8-group
  v = dpp_add<0x140>(v);   // row_mirror          : cross 8-groups in 16-row
  v += xhalf_f(v);         // cross half-wave
  return v;
}
__device__ inline float bcast0(float v) {  // lane0 broadcast (EXEC full, uniform flow)
  return __uint_as_float(__builtin_amdgcn_readfirstlane(__float_as_uint(v)));
}

// Lorentz point ops on a 64-vector held 2 comps/lane: a = comp[lane], b = comp[lane+32].
__device__ inline void dev_project(int lane, float& a, float& b) {
  float sa = fminf(fmaxf(a, -8.f), 8.f);
  float sb = fminf(fmaxf(b, -8.f), 8.f);
  float s  = wsum32(((lane == 0) ? 0.f : sa * sa) + sb * sb);
  float x0 = sqrtf(1.f + s + EPSF);
  a = (lane == 0) ? x0 : sa;
  b = sb;
}

__device__ inline void dev_logmap(int lane, float& a, float& b) {
  dev_project(lane, a, b);                 // log_map projects y first
  float y0 = bcast0(a);
  float xy = fminf(-y0, -(1.f + EPSF));
  float d  = fmaxf(acoshf(fmaxf(-xy, 1.f + EPSF)), 0.001f);
  float da = (lane == 0) ? (a + xy) : a;   // direc = y + xy*mu
  float db = b;
  float innr = wsum32(((lane == 0) ? -da * da : da * da) + db * db);
  float dn = sqrtf(fmaxf(innr, EPSF));
  float sc = d / (dn + EPSF);
  a = da * sc; b = db * sc;
}

__device__ inline void dev_expmap(int lane, float& a, float& b) {
  float vx = -bcast0(a);                   // inner(v, mu)
  float ta = (lane == 0) ? (a + vx) : a;   // vt = v + vx*mu
  float tb = b;
  float vns = fmaxf(wsum32(((lane == 0) ? -ta * ta : ta * ta) + tb * tb), 0.f);
  float vn  = fminf(sqrtf(vns + EPSF), 5.f);
  float ch  = coshf(vn);
  float sh  = sinhf(vn) / (vn + EPSF);
  a = ((lane == 0) ? ch : 0.f) + sh * ta;
  b = sh * tb;
  dev_project(lane, a, b);                 // exp_map ends with project
}

// ---------------- f32 -> f16 convert ----------------
__global__ __launch_bounds__(256) void cvt_f16_kernel(
    const float* __restrict__ src, _Float16* __restrict__ dst, int n) {
  int i = blockIdx.x * 256 + threadIdx.x;
  if (i < n) dst[i] = (_Float16)src[i];
}

// -------- WMMA GEMM (transposed tiles): out[tok][outcol] from C[m=outcol][n=tok] --------
// out[M=tokens, N] = Act[M,K] @ W[N,K]^T ; A-op = W rows, B-op = Act rows (contiguous).
template <int OUT_HALF>
__global__ __launch_bounds__(128) void gemm_kernel(
    const _Float16* __restrict__ W, const _Float16* __restrict__ Act,
    const float* __restrict__ bias, void* __restrict__ outp,
    int N, int K, int gelu) {
  int lane = threadIdx.x & 31;
  int wid  = threadIdx.x >> 5;
  int lo = lane & 15, hi = lane >> 4;
  int ntn  = N >> 4;
  int tile = blockIdx.x * 4 + wid;
  int tm = (tile / ntn) << 4;        // token tile
  int tn = (tile % ntn) << 4;        // outcol tile
  v8f acc = {};
  const _Float16* wrow = W   + (size_t)(tn + lo) * K + hi * 16;
  const _Float16* acr  = Act + (size_t)(tm + lo) * K + hi * 16;
  for (int kt = 0; kt < K; kt += 32) {
    v8h w0 = *(const v8h*)(wrow + kt);
    v8h w1 = *(const v8h*)(wrow + kt + 8);
    v16h bv = *(const v16h*)(acr + kt);
    acc = wmma16(cat88(w0, w1), bv, acc);
  }
  // lane holds outcols tn+hi*8+r (r=0..7) for token tm+lo -> contiguous store
  int token = tm + lo;
  int nbase = tn + hi * 8;
  float vr[8];
#pragma unroll
  for (int r = 0; r < 8; ++r) {
    float v = acc[r] + (bias ? bias[nbase + r] : 0.f);
    if (gelu) v = 0.5f * v * (1.f + erff(v * 0.7071067811865475f));
    vr[r] = v;
  }
  if (OUT_HALF) {
    v8h ov;
#pragma unroll
    for (int r = 0; r < 8; ++r) ov[r] = (_Float16)vr[r];
    *(v8h*)((_Float16*)outp + (size_t)token * N + nbase) = ov;
  } else {
    float4 s0 = make_float4(vr[0], vr[1], vr[2], vr[3]);
    float4 s1 = make_float4(vr[4], vr[5], vr[6], vr[7]);
    float* op = (float*)outp + (size_t)token * N + nbase;
    *(float4*)op       = s0;
    *(float4*)(op + 4) = s1;
  }
}

// ---------------- QKV projection (transposed) with fused ph epilogue ----------------
// mode 0: Q (ph + time-flip + 1/sqrt(DH) folded) -> f16 [B,H,T,DH]
// mode 1: K (ph)                                 -> f16 [B,H,T,DH]
// mode 2: V (raw, transposed)                    -> f16 [B,H,DH,T]
__global__ __launch_bounds__(128) void qkv_kernel(
    const _Float16* __restrict__ Act, const _Float16* __restrict__ W,
    _Float16* __restrict__ out, int mode) {
  int lane = threadIdx.x & 31;
  int wid  = threadIdx.x >> 5;
  int lo = lane & 15, hi = lane >> 4;
  int tile = blockIdx.x * 4 + wid;   // ntn = 4
  int tm = (tile >> 2) << 4;         // token tile
  int tn = (tile & 3) << 4;          // head tile (outcols)
  v8f acc = {};
  const _Float16* wrow = W   + (size_t)(tn + lo) * Dc + hi * 16;
  const _Float16* acr  = Act + (size_t)(tm + lo) * Dc + hi * 16;
#pragma unroll
  for (int kt = 0; kt < Dc; kt += 32) {
    v8h w0 = *(const v8h*)(wrow + kt);
    v8h w1 = *(const v8h*)(wrow + kt + 8);
    v16h bv = *(const v16h*)(acr + kt);
    acc = wmma16(cat88(w0, w1), bv, acc);
  }
  int token = tm + lo;
  int b = token >> 11, t = token & (Tc - 1);
  int head = tn >> 4;
  if (mode == 2) {                   // V^T: d = r+hi*8 for token t (strided in d)
#pragma unroll
    for (int r = 0; r < 8; ++r)
      out[((size_t)(b * NHc + head) * DHc + r + hi * 8) * Tc + t] = (_Float16)acc[r];
  } else {
    // per-head hyperboloid: sum clip^2 over d=1..15 (in-lane + 1 cross-half swap)
    float cl[8];
    float s = 0.f;
#pragma unroll
    for (int r = 0; r < 8; ++r) {
      cl[r] = fminf(fmaxf(acc[r], -5.f), 5.f);
      float c = cl[r] * cl[r];
      if (hi == 0 && r == 0) c = 0.f;   // exclude time comp
      s += c;
    }
    s += xhalf_f(s);
    float x0 = sqrtf(1.f + s + 1e-6f);
    v8h ov;
#pragma unroll
    for (int r = 0; r < 8; ++r) {
      float o = cl[r];
      if (hi == 0 && r == 0) o = x0;
      if (mode == 0) {                  // qm flip + 1/sqrt(DH)=0.25 folded
        if (hi == 0 && r == 0) o = -o;
        o *= 0.25f;
      }
      ov[r] = (_Float16)o;
    }
    *(v8h*)(out + ((size_t)(b * NHc + head) * Tc + t) * DHc + hi * 8) = ov;
  }
}

// ---------------- flash attention: 1 wave per 16-query tile (transposed S) ----------
// No LDS: P^T B-operand is each lane's own P values (keys hi*8+0..7), pad zeros.
__global__ __launch_bounds__(128) void attn_kernel(
    const _Float16* __restrict__ Q, const _Float16* __restrict__ K,
    const _Float16* __restrict__ Vt, _Float16* __restrict__ out) {
  int lane = threadIdx.x & 31;
  int wid  = threadIdx.x >> 5;
  int lo = lane & 15, hi = lane >> 4;
  int blk = blockIdx.x * 4 + wid;
  int qt = blk & (Tc / 16 - 1);
  int bh = blk / (Tc / 16);
  const _Float16* Qp = Q  + ((size_t)bh * Tc + qt * 16) * DHc;
  const _Float16* Kp = K  + (size_t)bh * Tc * DHc;
  const _Float16* Vp = Vt + (size_t)bh * DHc * Tc;
  v8h z8 = {};

  // B-op: Q columns (d contraction, pad 16->32): lane n=q=lo holds d=hi*8+0..7
  v16h bq = cat88(*(const v8h*)(Qp + lo * DHc + hi * 8), z8);

  float mi = -1e30f, li = 0.f;       // per-query stats (query = lo)
  v8f accOt = {};                    // O^T: lane holds O[d=r+hi*8][q=lo]

  for (int kt = 0; kt < Tc / 16; ++kt) {
    // A-op: K tile rows (16 keys x d16 pad32): row m=key=lo, halves d=hi*8+0..7
    v16h ak = cat88(*(const v8h*)(Kp + (size_t)(kt * 16 + lo) * DHc + hi * 8), z8);
    // A-op for PV: V^T rows (d x key16 pad32): row m=d=lo, halves key=hi*8+0..7
    v16h av = cat88(*(const v8h*)(Vp + (size_t)lo * Tc + kt * 16 + hi * 8), z8);
    v8f zc = {};
    v8f s = wmma16(ak, bq, zc);      // S^T[key=r+hi*8][q=lo]

    float mx = s[0];
#pragma unroll
    for (int r = 1; r < 8; ++r) mx = fmaxf(mx, s[r]);
    mx = fmaxf(mx, xhalf_f(mx));
    float mn = fmaxf(mi, mx);
    float rs = 0.f;
    union { v8h h; unsigned u[4]; } pk, po;
#pragma unroll
    for (int r = 0; r < 8; ++r) {
      float pv = __expf(s[r] - mn);
      rs += pv;
      pk.h[r] = (_Float16)pv;
    }
    rs += xhalf_f(rs);
    float co = __expf(mi - mn);
    li = li * co + rs;
    mi = mn;
#pragma unroll
    for (int r = 0; r < 8; ++r) accOt[r] *= co;
    // Assemble P^T B-operand: lane(q=lo) holds own keys hi*8+0..7, pad zeros.
#pragma unroll
    for (int j = 0; j < 4; ++j) po.u[j] = xhalf_u32(pk.u[j]);
    v16h bp = cat88(pk.h, z8);
    accOt = wmma16(av, bp, accOt);
  }
  // lane: token qt*16+lo, d = h*16 + hi*8 + r -> contiguous v8h store
  int b = bh >> 2;
  int h = bh & 3;
  float inv = 1.f / li;
  v8h ov;
#pragma unroll
  for (int r = 0; r < 8; ++r) ov[r] = (_Float16)(accOt[r] * inv);
  *(v8h*)(out + ((size_t)b * Tc + qt * 16 + lo) * Dc + h * DHc + hi * 8) = ov;
}

// ---------------- elementwise Lorentz kernels (1 wave / token) ----------------
__global__ __launch_bounds__(256) void embed_kernel(
    const float* __restrict__ x, const float* __restrict__ eW,
    const float* __restrict__ eB, float* __restrict__ h) {
  int tok  = blockIdx.x * 8 + (threadIdx.x >> 5);
  int lane = threadIdx.x & 31;
  const float* xr = x + (size_t)tok * 6;
  float xv[6];
#pragma unroll
  for (int j = 0; j < 6; ++j) xv[j] = xr[j];
  float a = eB[lane], b = eB[lane + 32];
#pragma unroll
  for (int j = 0; j < 6; ++j) {
    a += xv[j] * eW[lane * 6 + j];
    b += xv[j] * eW[(lane + 32) * 6 + j];
  }
  dev_project(lane, a, b);                    // h = project(x@W^T+b)
  dev_logmap(lane, a, b);                     // v = log_map(mu,h)
  int t = tok & (Tc - 1);
  if (lane == 0) a += ((float)t / (float)Tc) * 0.1f;
  dev_expmap(lane, a, b);
  dev_project(lane, a, b);                    // h = project(exp_map(mu,v))
  h[(size_t)tok * Dc + lane]      = a;
  h[(size_t)tok * Dc + lane + 32] = b;
}

__global__ __launch_bounds__(256) void logmap_kernel(
    const float* __restrict__ h, float* __restrict__ vlog,
    _Float16* __restrict__ vlogh) {
  int tok  = blockIdx.x * 8 + (threadIdx.x >> 5);
  int lane = threadIdx.x & 31;
  float a = h[(size_t)tok * Dc + lane];
  float b = h[(size_t)tok * Dc + lane + 32];
  dev_logmap(lane, a, b);
  vlog[(size_t)tok * Dc + lane]       = a;
  vlog[(size_t)tok * Dc + lane + 32]  = b;
  vlogh[(size_t)tok * Dc + lane]      = (_Float16)a;
  vlogh[(size_t)tok * Dc + lane + 32] = (_Float16)b;
}

// lin -> project(exp_map) -> log_map -> +vlog -> LN -> project(exp_map) -> h [-> vlog]
__global__ __launch_bounds__(256) void merge_kernel(
    const float* __restrict__ lin, float* __restrict__ vlog,
    _Float16* __restrict__ vlogh, float* __restrict__ h,
    const float* __restrict__ g, const float* __restrict__ be, int write_vlog) {
  int tok  = blockIdx.x * 8 + (threadIdx.x >> 5);
  int lane = threadIdx.x & 31;
  float a = lin[(size_t)tok * Dc + lane];
  float b = lin[(size_t)tok * Dc + lane + 32];
  dev_expmap(lane, a, b);
  dev_project(lane, a, b);                    // sub-module output point
  dev_logmap(lane, a, b);                     // log_map of sub-module output
  a += vlog[(size_t)tok * Dc + lane];
  b += vlog[(size_t)tok * Dc + lane + 32];
  float m = wsum32(a + b) * (1.f / 64.f);     // layernorm over 64
  float ca = a - m, cb = b - m;
  float var = wsum32(ca * ca + cb * cb) * (1.f / 64.f);
  float inv = rsqrtf(var + 1e-5f);
  a = ca * inv * g[lane]      + be[lane];
  b = cb * inv * g[lane + 32] + be[lane + 32];
  dev_expmap(lane, a, b);
  dev_project(lane, a, b);
  h[(size_t)tok * Dc + lane]      = a;
  h[(size_t)tok * Dc + lane + 32] = b;
  if (write_vlog) {                           // FFN input = log_map(mu, x_mid)
    dev_logmap(lane, a, b);
    vlog[(size_t)tok * Dc + lane]       = a;
    vlog[(size_t)tok * Dc + lane + 32]  = b;
    vlogh[(size_t)tok * Dc + lane]      = (_Float16)a;
    vlogh[(size_t)tok * Dc + lane + 32] = (_Float16)b;
  }
}

// ---------------- trajectory head ----------------
__global__ __launch_bounds__(128) void traj_kernel(
    const float* __restrict__ h, const float* __restrict__ trW,
    const float* __restrict__ trb, float* __restrict__ out) {
  __shared__ float e[64];
  int b = blockIdx.x, tid = threadIdx.x;
  if (tid < 64) {
    float v = h[((size_t)b * Tc + (Tc - 1)) * Dc + tid];
    e[tid] = (tid == 0) ? v : fminf(fmaxf(v, -8.f), 8.f);
  }
  __syncthreads();
  if (tid == 0) {
    float s = 0.f;
    for (int c = 1; c < 64; ++c) s += e[c] * e[c];
    e[0] = sqrtf(1.f + s + EPSF);
  }
  __syncthreads();
  if (tid < 120) {
    float acc = trb[tid];
    for (int c = 0; c < 64; ++c) acc += e[c] * trW[tid * 64 + c];
    out[b * 120 + tid] = acc;
  }
}

// ---------------- driver ----------------
extern "C" void kernel_launch(void* const* d_in, const int* in_sizes, int n_in,
                              void* d_out, int out_size, void* d_ws, size_t ws_size,
                              hipStream_t stream) {
  (void)in_sizes; (void)n_in; (void)out_size; (void)ws_size;
  const float* x    = (const float*)d_in[0];
  const float* eW   = (const float*)d_in[1];
  const float* eB   = (const float*)d_in[2];
  const float* Wq   = (const float*)d_in[3];
  const float* Wk   = (const float*)d_in[4];
  const float* Wv   = (const float*)d_in[5];
  const float* Wo   = (const float*)d_in[6];
  const float* fc1W = (const float*)d_in[7];
  const float* fc1b = (const float*)d_in[8];
  const float* fc2W = (const float*)d_in[9];
  const float* fc2b = (const float*)d_in[10];
  const float* ln1g = (const float*)d_in[11];
  const float* ln1b = (const float*)d_in[12];
  const float* ln2g = (const float*)d_in[13];
  const float* ln2b = (const float*)d_in[14];
  const float* trW  = (const float*)d_in[15];
  const float* trb  = (const float*)d_in[16];
  float* out = (float*)d_out;

  // workspace carving (256B aligned)
  char* p = (char*)d_ws;
  auto alloc = [&](size_t bytes) {
    char* r = p;
    p += (bytes + 255) & ~(size_t)255;
    return r;
  };
  size_t n64 = (size_t)BTc * Dc;
  float*    h     = (float*)alloc(n64 * 4);
  float*    vlog  = (float*)alloc(n64 * 4);
  float*    tmp   = (float*)alloc(n64 * 4);
  _Float16* vlogh = (_Float16*)alloc(n64 * 2);
  _Float16* qh    = (_Float16*)alloc(n64 * 2);
  _Float16* kh    = (_Float16*)alloc(n64 * 2);
  _Float16* vth   = (_Float16*)alloc(n64 * 2);
  _Float16* aoh   = (_Float16*)alloc(n64 * 2);
  _Float16* ffnh  = (_Float16*)alloc((size_t)BTc * FFc * 2);
  _Float16* wqh   = (_Float16*)alloc((size_t)3 * Dc * Dc * 2);
  _Float16* wkh   = (_Float16*)alloc((size_t)3 * Dc * Dc * 2);
  _Float16* wvh   = (_Float16*)alloc((size_t)3 * Dc * Dc * 2);
  _Float16* woh   = (_Float16*)alloc((size_t)3 * Dc * Dc * 2);
  _Float16* f1h   = (_Float16*)alloc((size_t)3 * FFc * Dc * 2);
  _Float16* f2h   = (_Float16*)alloc((size_t)3 * Dc * FFc * 2);

  const int nW  = 3 * Dc * Dc;    // 12288
  const int nF  = 3 * FFc * Dc;   // 49152
  cvt_f16_kernel<<<(nW + 255) / 256, 256, 0, stream>>>(Wq, wqh, nW);
  cvt_f16_kernel<<<(nW + 255) / 256, 256, 0, stream>>>(Wk, wkh, nW);
  cvt_f16_kernel<<<(nW + 255) / 256, 256, 0, stream>>>(Wv, wvh, nW);
  cvt_f16_kernel<<<(nW + 255) / 256, 256, 0, stream>>>(Wo, woh, nW);
  cvt_f16_kernel<<<(nF + 255) / 256, 256, 0, stream>>>(fc1W, f1h, nF);
  cvt_f16_kernel<<<(nF + 255) / 256, 256, 0, stream>>>(fc2W, f2h, nF);

  const int ewBlocks   = BTc / 8;                      // 2048 (wave/token)
  const int qkvBlocks  = (BTc / 16) * (Dc / 16) / 4;   // 1024 (4 waves/block)
  const int woBlocks   = qkvBlocks;
  const int fc1Blocks  = (BTc / 16) * (FFc / 16) / 4;  // 4096
  const int attnBlocks = Bc * NHc * (Tc / 16) / 4;     // 1024 (4 waves/block)

  embed_kernel<<<ewBlocks, 256, 0, stream>>>(x, eW, eB, h);
  for (int l = 0; l < 3; ++l) {
    logmap_kernel<<<ewBlocks, 256, 0, stream>>>(h, vlog, vlogh);
    qkv_kernel<<<qkvBlocks, 128, 0, stream>>>(vlogh, wqh + (size_t)l * Dc * Dc, qh, 0);
    qkv_kernel<<<qkvBlocks, 128, 0, stream>>>(vlogh, wkh + (size_t)l * Dc * Dc, kh, 1);
    qkv_kernel<<<qkvBlocks, 128, 0, stream>>>(vlogh, wvh + (size_t)l * Dc * Dc, vth, 2);
    attn_kernel<<<attnBlocks, 128, 0, stream>>>(qh, kh, vth, aoh);
    gemm_kernel<0><<<woBlocks, 128, 0, stream>>>(woh + (size_t)l * Dc * Dc, aoh,
                                                 nullptr, tmp, Dc, Dc, 0);
    merge_kernel<<<ewBlocks, 256, 0, stream>>>(tmp, vlog, vlogh, h,
                                               ln1g + l * Dc, ln1b + l * Dc, 1);
    gemm_kernel<1><<<fc1Blocks, 128, 0, stream>>>(f1h + (size_t)l * FFc * Dc, vlogh,
                                                  fc1b + l * FFc, ffnh, FFc, Dc, 1);
    gemm_kernel<0><<<woBlocks, 128, 0, stream>>>(f2h + (size_t)l * Dc * FFc, ffnh,
                                                 fc2b + l * Dc, tmp, Dc, FFc, 0);
    merge_kernel<<<ewBlocks, 256, 0, stream>>>(tmp, vlog, vlogh, h,
                                               ln2g + l * Dc, ln2b + l * Dc, 0);
  }
  traj_kernel<<<Bc, 128, 0, stream>>>(h, trW, trb, out);
}
